// DLiNOSSLayer_85976655331766
// MI455X (gfx1250) — matrix operand order repaired and checked by reading, not compiled, not run on hardware
//
#include <hip/hip_runtime.h>

// ---------------------------------------------------------------------------
// DLiNOSS layer for MI455X (gfx1250, wave32, WMMA).
//   k_precompute_A : A = mag * exp(i*omega)
//   k_gemm_in      : WMMA f16->f32 GEMMs (W_g, B_re, B_im), fused sigmoid gate
//                    epilogue -> interleaved float4 stream (ar, ai, br, bi)
//   k_scan_pass1/2/3: chunked complex linear-recurrence scan over L
//                    (h -> interleaved float2 stream; h_last -> d_out tail)
//   k_gemm_out     : WMMA GEMM Re{C*h} + x*D_skip -> y
// LDS weight tiles are pre-swizzled into WMMA B-fragment layout so each lane
// fetches its 16 halves with two ds_load_b128 (no scalar ds_load_u16 gather).
// ---------------------------------------------------------------------------

#define H_      16
#define DH_     64
#define S_      64
#define SFAST_  48
#define BATCH_  4
#define L_      2048
#define D_      1024
#define ROWS_   (BATCH_ * L_)              // 8192
#define NTOT_   ((long)ROWS_ * H_ * S_)    // 8,388,608 elems per stream
#define CHUNK_  256
#define NCHUNK_ (L_ / CHUNK_)              // 8
#define NSEQ_   (BATCH_ * H_ * S_)         // 4096 independent scan chains
#define Y_SIZE_ ((long)ROWS_ * D_)         // 8,388,608

typedef __attribute__((ext_vector_type(16))) _Float16 v16h;
typedef __attribute__((ext_vector_type(8)))  float    v8f;

union HalfPack { _Float16 h[2]; unsigned u; };

__device__ __forceinline__ float fast_sigmoid(float z) {
  // v_exp_f32 + v_rcp_f32 (avoids the IEEE div_scale/fma correction chain)
  return __builtin_amdgcn_rcpf(1.0f + __expf(-z));
}

// Convert two 8-float contiguous runs (k = base+0..7 and base+16..23) into the
// ISA 16-bit A-fragment register layout for one 16x32 k-step.
__device__ __forceinline__ v16h load_afrag(const float* kwin, int basek) {
  const float4* p0 = (const float4*)(kwin + basek);
  const float4* p1 = (const float4*)(kwin + 16 + basek);
  float4 a = p0[0], b = p0[1], c = p1[0], d = p1[1];
  v16h f;
  f[0]  = (_Float16)a.x; f[1]  = (_Float16)a.y; f[2]  = (_Float16)a.z; f[3]  = (_Float16)a.w;
  f[4]  = (_Float16)b.x; f[5]  = (_Float16)b.y; f[6]  = (_Float16)b.z; f[7]  = (_Float16)b.w;
  f[8]  = (_Float16)c.x; f[9]  = (_Float16)c.y; f[10] = (_Float16)c.z; f[11] = (_Float16)c.w;
  f[12] = (_Float16)d.x; f[13] = (_Float16)d.y; f[14] = (_Float16)d.z; f[15] = (_Float16)d.w;
  return f;
}

// ---------------------------------------------------------------------------
// Phase 0: A[h,s] = mag * exp(i*omega)
// ---------------------------------------------------------------------------
__global__ void k_precompute_A(const float* __restrict__ log_alpha,
                               const float* __restrict__ omega,
                               float* __restrict__ Are,
                               float* __restrict__ Aim) {
  int i = blockIdx.x * blockDim.x + threadIdx.x;
  if (i >= H_ * S_) return;
  int h = i / S_, s = i % S_;
  float mag = 1.0f;
  if (s < SFAST_) {
    float la = log_alpha[h * SFAST_ + s];
    float sp = (la > 20.f) ? la : log1pf(__expf(la));   // softplus
    mag = fminf(__expf(-sp), 0.999f);
  }
  float om = omega[i];
  Are[i] = mag * __cosf(om);
  Aim[i] = mag * __sinf(om);
}

// ---------------------------------------------------------------------------
// Phase 1: per-head GEMMs (M=8192, N=64, K=64) with fused gate epilogue.
// Grid: (ROWS_/128, H_), 256 threads = 8 waves, 1 wave per 16-row tile.
// LDS: fragment-swizzled weights [mat][kstep][ntile][lane][8 x u32] = 24 KB.
// ---------------------------------------------------------------------------
__global__ __launch_bounds__(256) void k_gemm_in(
    const float* __restrict__ x,   const float* __restrict__ Wg,
    const float* __restrict__ Bre, const float* __restrict__ Bim,
    const float* __restrict__ bg,  const float* __restrict__ Are,
    const float* __restrict__ Aim, float4* __restrict__ ab) {
  __shared__ __attribute__((aligned(32))) unsigned lwp[3][2][4][32][8];
  const int h   = blockIdx.y;
  const int tid = threadIdx.x;

  const float* mats[3] = { Wg  + (long)h * S_ * DH_,
                           Bre + (long)h * S_ * DH_,
                           Bim + (long)h * S_ * DH_ };
  // Stage weights: element (s,d) -> B-fragment slot. k=d, n=s.
  for (int i = tid; i < 3 * S_ * DH_ / 2; i += 256) {
    int mm = i / (S_ * DH_ / 2);
    int r  = i % (S_ * DH_ / 2);
    int s  = r / (DH_ / 2);
    int d  = (r % (DH_ / 2)) * 2;                       // even k
    float2 w = *(const float2*)(mats[mm] + s * DH_ + d);
    int ks   = d >> 5;
    int kk   = d & 31;
    int lane = (((kk >> 3) & 1) << 4) | (s & 15);
    HalfPack p; p.h[0] = (_Float16)w.x; p.h[1] = (_Float16)w.y;
    lwp[mm][ks][s >> 4][lane][((kk & 7) | ((kk >> 4) << 3)) >> 1] = p.u;
  }
  __syncthreads();

  const int wave  = tid >> 5;
  const int lane  = tid & 31;
  const int row0  = (blockIdx.x * 8 + wave) * 16;
  const int mA    = lane & 15;
  const int basek = (lane >> 4) * 8;

  v16h afrag[2];
  const float* xrow = x + (long)(row0 + mA) * D_ + h * DH_;
  afrag[0] = load_afrag(xrow, basek);
  afrag[1] = load_afrag(xrow + 32, basek);

  const int n    = lane & 15;
  const int mrow = (lane >> 4) * 8;

  for (int n0 = 0; n0 < 4; ++n0) {
    v8f acc[3];
    for (int mm = 0; mm < 3; ++mm) {
      v8f c = {};
      for (int ks = 0; ks < 2; ++ks) {
        v16h bfrag = *(const v16h*)&lwp[mm][ks][n0][lane][0];  // 2x ds_load_b128
        c = __builtin_amdgcn_wmma_f32_16x16x32_f16(
                false, afrag[ks], false, bfrag, (short)0, c, false, false);
      }
      acc[mm] = c;
    }
    int s = n0 * 16 + n;
    float are = Are[h * S_ + s], aim = Aim[h * S_ + s];
    float bgs = bg[h * S_ + s];
    for (int j = 0; j < 8; ++j) {
      int  row = row0 + mrow + j;
      long idx = ((long)row * H_ + h) * S_ + s;
      float g = fast_sigmoid(acc[0][j] + bgs);
      float ar, ai;
      if (s < SFAST_) { ar = g * are; ai = g * aim; }
      else            { ar = are;     ai = aim;     }
      float omg = 1.f - g;
      ab[idx] = make_float4(ar, ai, omg * acc[1][j], omg * acc[2][j]);
    }
  }
}

// ---------------------------------------------------------------------------
// Phase 2: chunked scan of h[t] = alpha[t]*h[t-1] + beta[t] (complex diag).
// Outer loop kept rolled (small I$); inner 8 steps unrolled so the 8
// independent b128 loads issue ahead of the dependent FMA chain.
// ---------------------------------------------------------------------------
__global__ void k_scan_pass1(const float4* __restrict__ ab,
                             float4* __restrict__ carry) {
  int tid = blockIdx.x * blockDim.x + threadIdx.x;
  if (tid >= NSEQ_ * NCHUNK_) return;
  int s  = tid % S_;
  int hh = (tid / S_) % H_;
  int ck = (tid / (S_ * H_)) % NCHUNK_;
  int b  = tid / (S_ * H_ * NCHUNK_);
  float Ar = 1.f, Ai = 0.f, Br = 0.f, Bi = 0.f;
  long base = (((long)b * L_ + (long)ck * CHUNK_) * H_ + hh) * S_ + s;
#pragma unroll 1
  for (int t0 = 0; t0 < CHUNK_; t0 += 8) {
#pragma unroll
    for (int u = 0; u < 8; ++u) {
      float4 v = ab[base + (long)(t0 + u) * (H_ * S_)];   // one b128
      float nAr = v.x * Ar - v.y * Ai;
      float nAi = v.x * Ai + v.y * Ar;
      float nBr = v.x * Br - v.y * Bi + v.z;
      float nBi = v.x * Bi + v.y * Br + v.w;
      Ar = nAr; Ai = nAi; Br = nBr; Bi = nBi;
    }
  }
  carry[((b * H_ + hh) * S_ + s) * NCHUNK_ + ck] = make_float4(Ar, Ai, Br, Bi);
}

__global__ void k_scan_pass2(const float4* __restrict__ carry,
                             const float* __restrict__ h0,
                             float2* __restrict__ st) {
  int cseq = blockIdx.x * blockDim.x + threadIdx.x;
  if (cseq >= NSEQ_) return;
  float hr = h0[2 * cseq], hi = h0[2 * cseq + 1];      // complex64 interleaved
  for (int ck = 0; ck < NCHUNK_; ++ck) {
    int cidx = cseq * NCHUNK_ + ck;
    st[cidx] = make_float2(hr, hi);                    // state entering chunk
    float4 c = carry[cidx];
    float nr = c.x * hr - c.y * hi + c.z;
    float ni = c.x * hi + c.y * hr + c.w;
    hr = nr; hi = ni;
  }
}

__global__ void k_scan_pass3(const float4* __restrict__ ab,
                             float2* __restrict__ hc,
                             const float2* __restrict__ st,
                             float* __restrict__ hlast) {
  int tid = blockIdx.x * blockDim.x + threadIdx.x;
  if (tid >= NSEQ_ * NCHUNK_) return;
  int s  = tid % S_;
  int hh = (tid / S_) % H_;
  int ck = (tid / (S_ * H_)) % NCHUNK_;
  int b  = tid / (S_ * H_ * NCHUNK_);
  int cseq = (b * H_ + hh) * S_ + s;
  float2 h = st[cseq * NCHUNK_ + ck];
  float hr = h.x, hi = h.y;
  long base = (((long)b * L_ + (long)ck * CHUNK_) * H_ + hh) * S_ + s;
#pragma unroll 1
  for (int t0 = 0; t0 < CHUNK_; t0 += 8) {
#pragma unroll
    for (int u = 0; u < 8; ++u) {
      long idx = base + (long)(t0 + u) * (H_ * S_);
      float4 v = ab[idx];
      float nr = v.x * hr - v.y * hi + v.z;
      float ni = v.x * hi + v.y * hr + v.w;
      hr = nr; hi = ni;
      hc[idx] = make_float2(hr, hi);          // one b64 store
    }
  }
  if (ck == NCHUNK_ - 1) { hlast[2 * cseq] = hr; hlast[2 * cseq + 1] = hi; }
}

// ---------------------------------------------------------------------------
// Phase 3: y = Re{C*h} + x*D_skip, Re{} fused via h_re x C_re + h_im x (-C_im)
// accumulated into one f32 WMMA chain.
// ---------------------------------------------------------------------------
__global__ __launch_bounds__(256) void k_gemm_out(
    const float2* __restrict__ hc,
    const float* __restrict__ Cre, const float* __restrict__ Cim,
    const float* __restrict__ x,   const float* __restrict__ Dskip,
    float* __restrict__ y) {
  __shared__ __attribute__((aligned(32))) unsigned lcp[2][2][4][32][8];  // 16 KB
  const int h   = blockIdx.y;
  const int tid = threadIdx.x;
  // Stage C: element (d, s): k=s, n=d. Matrix 1 holds -C_im.
  for (int i = tid; i < 2 * DH_ * S_ / 2; i += 256) {
    int mm = i / (DH_ * S_ / 2);
    int r  = i % (DH_ * S_ / 2);
    int d  = r / (S_ / 2);
    int s  = (r % (S_ / 2)) * 2;                        // even k
    const float* srcp = (mm == 0 ? Cre : Cim) + ((long)h * DH_ + d) * S_ + s;
    float2 w = *(const float2*)srcp;
    if (mm) { w.x = -w.x; w.y = -w.y; }
    int ks   = s >> 5;
    int kk   = s & 31;
    int lane = (((kk >> 3) & 1) << 4) | (d & 15);
    HalfPack p; p.h[0] = (_Float16)w.x; p.h[1] = (_Float16)w.y;
    lcp[mm][ks][d >> 4][lane][((kk & 7) | ((kk >> 4) << 3)) >> 1] = p.u;
  }
  __syncthreads();

  const int wave  = tid >> 5;
  const int lane  = tid & 31;
  const int row0  = (blockIdx.x * 8 + wave) * 16;
  const int mA    = lane & 15;
  const int basek = (lane >> 4) * 8;

  // h fragments: interleaved (re,im); 4x b128 per 8-element run.
  v16h afr[2], afi[2];
  const float2* hrow = hc + ((long)(row0 + mA) * H_ + h) * S_;
  for (int ks = 0; ks < 2; ++ks) {
    const float4* p0 = (const float4*)(hrow + ks * 32 + basek);
    const float4* p1 = (const float4*)(hrow + ks * 32 + 16 + basek);
#pragma unroll
    for (int q = 0; q < 4; ++q) {
      float4 a = p0[q];
      afr[ks][2 * q]     = (_Float16)a.x;  afi[ks][2 * q]     = (_Float16)a.y;
      afr[ks][2 * q + 1] = (_Float16)a.z;  afi[ks][2 * q + 1] = (_Float16)a.w;
      float4 b = p1[q];
      afr[ks][8 + 2 * q]     = (_Float16)b.x;  afi[ks][8 + 2 * q]     = (_Float16)b.y;
      afr[ks][8 + 2 * q + 1] = (_Float16)b.z;  afi[ks][8 + 2 * q + 1] = (_Float16)b.w;
    }
  }

  const int n    = lane & 15;
  const int mrow = (lane >> 4) * 8;

  for (int n0 = 0; n0 < 4; ++n0) {
    v8f c = {};
    for (int ks = 0; ks < 2; ++ks) {
      v16h bre = *(const v16h*)&lcp[0][ks][n0][lane][0];
      v16h bim = *(const v16h*)&lcp[1][ks][n0][lane][0];
      c = __builtin_amdgcn_wmma_f32_16x16x32_f16(
              false, afr[ks], false, bre, (short)0, c, false, false);
      c = __builtin_amdgcn_wmma_f32_16x16x32_f16(
              false, afi[ks], false, bim, (short)0, c, false, false);
    }
    for (int j = 0; j < 8; ++j) {
      int  row  = row0 + mrow + j;
      int  d    = n0 * 16 + n;
      long yidx = (long)row * D_ + h * DH_ + d;
      y[yidx] = c[j] + x[yidx] * Dskip[h * DH_ + d];
    }
  }
}

// ---------------------------------------------------------------------------
extern "C" void kernel_launch(void* const* d_in, const int* in_sizes, int n_in,
                              void* d_out, int out_size, void* d_ws, size_t ws_size,
                              hipStream_t stream) {
  (void)in_sizes; (void)n_in; (void)out_size; (void)ws_size;
  const float* x         = (const float*)d_in[0];
  const float* log_alpha = (const float*)d_in[1];
  const float* omega     = (const float*)d_in[2];
  const float* B_re      = (const float*)d_in[3];
  const float* B_im      = (const float*)d_in[4];
  const float* C_re      = (const float*)d_in[5];
  const float* C_im      = (const float*)d_in[6];
  const float* D_skip    = (const float*)d_in[7];
  const float* W_g       = (const float*)d_in[8];
  const float* b_g       = (const float*)d_in[9];
  const float* h0        = (const float*)d_in[10];   // complex64 interleaved
  float* out = (float*)d_out;

  // Workspace: A (8 KB) + ab float4 stream (134 MB) + h float2 stream (67 MB)
  // + carries/states (768 KB). ab/h largely L2-resident between passes.
  float*  ws    = (float*)d_ws;
  float*  Are   = ws;
  float*  Aim   = ws + H_ * S_;
  float4* ab    = (float4*)(ws + 2 * H_ * S_);
  float2* hc    = (float2*)((float*)ab + NTOT_ * 4);
  float4* carry = (float4*)((float*)hc + NTOT_ * 2);
  float2* st    = (float2*)((float*)carry + (long)NSEQ_ * NCHUNK_ * 4);

  k_precompute_A<<<dim3((H_ * S_ + 255) / 256), 256, 0, stream>>>(
      log_alpha, omega, Are, Aim);

  k_gemm_in<<<dim3(ROWS_ / 128, H_), 256, 0, stream>>>(
      x, W_g, B_re, B_im, b_g, Are, Aim, ab);

  k_scan_pass1<<<dim3(NSEQ_ * NCHUNK_ / 256), 256, 0, stream>>>(ab, carry);

  k_scan_pass2<<<dim3(NSEQ_ / 256), 256, 0, stream>>>(carry, h0, st);

  k_scan_pass3<<<dim3(NSEQ_ * NCHUNK_ / 256), 256, 0, stream>>>(
      ab, hc, st, out + Y_SIZE_);

  k_gemm_out<<<dim3(ROWS_ / 128, H_), 256, 0, stream>>>(
      hc, C_re, C_im, x, D_skip, out);
}
